// LSTM_70317204570620
// MI455X (gfx1250) — matrix operand order repaired and checked by reading, not compiled, not run on hardware
//
#include <hip/hip_runtime.h>
#include <hip/hip_bf16.h>

// ---------------------------------------------------------------------------
// LSTM on gfx1250 (MI455X), bf16 WMMA with fp32 accumulation.
//   Phase 1: convert X -> bf16; transpose+convert all weights to [N,K] bf16.
//   Phase 2: Xp = X @ [Wxi|Wxf|Wxo|Wxc] + b  (one big WMMA GEMM, 16384x4096)
//   Phase 3: 256 fused step kernels: G = H@[Whi|Whf|Who|Whc] + Xp[t], gates,
//            C/H update (serial critical path, weights L2-resident: 8MB bf16)
//   Phase 4: outputs = H_all @ Whq + bq     (one big WMMA GEMM, 16384x1024)
// K loops are software-pipelined: next fragment set is loaded while the
// current one feeds the WMMAs, hiding L2 latency inside the serial region.
// ---------------------------------------------------------------------------

typedef __attribute__((ext_vector_type(16))) __bf16 v16bf;
typedef __attribute__((ext_vector_type(8)))  float  v8f;

union Frag { uint4 u[2]; v16bf v; };
union Pack8 { unsigned short s[8]; uint4 u; };

__device__ __forceinline__ unsigned short f2bf(float f) {
    unsigned int u = __float_as_uint(f);
    u += 0x7FFFu + ((u >> 16) & 1u);            // round-to-nearest-even
    return (unsigned short)(u >> 16);
}
__device__ __forceinline__ float sigmoidf(float x) {
    return 1.0f / (1.0f + __expf(-x));
}
__device__ __forceinline__ float tanh_fast(float x) {
    return 1.0f - 2.0f / (1.0f + __expf(2.0f * x));   // safe at +/-inf
}

// ---- fp32 -> bf16, 8 elements per thread (vectorized b128 in / b128 out) ---
__global__ __launch_bounds__(256) void f32_to_bf16_kernel(
    const float* __restrict__ in, unsigned short* __restrict__ out, size_t n8)
{
    size_t i = (size_t)blockIdx.x * blockDim.x + threadIdx.x;
    if (i >= n8) return;
    const float4* p = (const float4*)in + i * 2;
    float4 a = p[0], b = p[1];
    Pack8 o;
    o.s[0]=f2bf(a.x); o.s[1]=f2bf(a.y); o.s[2]=f2bf(a.z); o.s[3]=f2bf(a.w);
    o.s[4]=f2bf(b.x); o.s[5]=f2bf(b.y); o.s[6]=f2bf(b.z); o.s[7]=f2bf(b.w);
    ((uint4*)out)[i] = o.u;
}

// ---- [K,N] fp32 -> [N,K] bf16 transpose via LDS 32x32 tile ----------------
__global__ void transpose_bf16_kernel(
    const float* __restrict__ in, unsigned short* __restrict__ out,
    int K, int N)
{
    __shared__ unsigned short tile[32][33];
    const int tx = threadIdx.x, ty = threadIdx.y;      // 32 x 8
    const int kt = blockIdx.y * 32, nt = blockIdx.x * 32;
    #pragma unroll
    for (int r = 0; r < 4; ++r) {
        int k = kt + ty + r * 8;
        tile[ty + r * 8][tx] = f2bf(in[(size_t)k * N + nt + tx]);
    }
    __syncthreads();
    #pragma unroll
    for (int r = 0; r < 4; ++r) {
        int n = nt + ty + r * 8;
        out[(size_t)n * K + kt + tx] = tile[tx][ty + r * 8];
    }
}

// ---- init: H0 -> bf16 + fp32 copies, C0 -> working C ----------------------
__global__ __launch_bounds__(256) void init_state_kernel(
    const float* __restrict__ H0, const float* __restrict__ C0,
    unsigned short* __restrict__ H0bf, float* __restrict__ C,
    float* __restrict__ Hf32, int n)
{
    int i = blockIdx.x * blockDim.x + threadIdx.x;
    if (i >= n) return;
    H0bf[i] = f2bf(H0[i]);
    C[i]    = C0[i];
    Hf32[i] = H0[i];
}

// ---- concat gate biases [4096] --------------------------------------------
__global__ __launch_bounds__(256) void build_bias_kernel(
    const float* __restrict__ bi, const float* __restrict__ bf_,
    const float* __restrict__ bo, const float* __restrict__ bc,
    float* __restrict__ bcat)
{
    int i = blockIdx.x * blockDim.x + threadIdx.x;   // 0..4095
    const float* src = (i < 1024) ? bi : (i < 2048) ? bf_ : (i < 3072) ? bo : bc;
    bcat[i] = src[i & 1023];
}

// ---- generic bf16 WMMA GEMM: out[M,N] = A[M,K] * BT[N,K]^T + bias[N] ------
// block = 256 threads = 8 waves; block tile 32(M) x 256(N);
// wave tile 16 x 64 (4 accumulators sharing one A fragment).
__global__ __launch_bounds__(256) void gemm_bf16_kernel(
    const unsigned short* __restrict__ A,
    const unsigned short* __restrict__ BT,
    const float* __restrict__ bias,
    float* __restrict__ out, int N, int K)
{
    const int lane  = threadIdx.x & 31;
    const int wave  = threadIdx.x >> 5;
    const int row0  = blockIdx.y * 32  + (wave & 1) * 16;
    const int col0  = blockIdx.x * 256 + (wave >> 1) * 64;
    const int mA    = lane & 15;
    const int khalf = lane >> 4;             // A: lanes 16-31 hold K {8..15,24..31}
    const int nB    = lane & 15;
    const int klo   = (lane >> 4) << 3;      // B: lanes 16-31 hold K {8..15,24..31}

    v8f acc[4] = {{}, {}, {}, {}};
    const unsigned short* Arow = A + (size_t)(row0 + mA) * K + khalf * 8;
    const unsigned short* Bp[4];
    #pragma unroll
    for (int c = 0; c < 4; ++c)
        Bp[c] = BT + (size_t)(col0 + c * 16 + nB) * K + klo;

    // prologue: fragment set for kb = 0
    Frag a_c, b_c[4];
    a_c.u[0] = *(const uint4*)(Arow);
    a_c.u[1] = *(const uint4*)(Arow + 16);
    #pragma unroll
    for (int c = 0; c < 4; ++c) {
        b_c[c].u[0] = *(const uint4*)(Bp[c]);
        b_c[c].u[1] = *(const uint4*)(Bp[c] + 16);
    }

    // steady state: issue loads for kb, compute on kb-32
    #pragma unroll 2
    for (int kb = 32; kb < K; kb += 32) {
        Frag a_n, b_n[4];
        a_n.u[0] = *(const uint4*)(Arow + kb);
        a_n.u[1] = *(const uint4*)(Arow + kb + 16);
        #pragma unroll
        for (int c = 0; c < 4; ++c) {
            b_n[c].u[0] = *(const uint4*)(Bp[c] + kb);
            b_n[c].u[1] = *(const uint4*)(Bp[c] + kb + 16);
        }
        #pragma unroll
        for (int c = 0; c < 4; ++c)
            acc[c] = __builtin_amdgcn_wmma_f32_16x16x32_bf16(
                false, a_c.v, false, b_c[c].v, (short)0, acc[c], false, false);
        a_c = a_n;
        #pragma unroll
        for (int c = 0; c < 4; ++c) b_c[c] = b_n[c];
    }
    // epilogue: final fragment set
    #pragma unroll
    for (int c = 0; c < 4; ++c)
        acc[c] = __builtin_amdgcn_wmma_f32_16x16x32_bf16(
            false, a_c.v, false, b_c[c].v, (short)0, acc[c], false, false);

    #pragma unroll
    for (int v = 0; v < 8; ++v) {
        int m = row0 + v + ((lane >> 4) << 3);
        #pragma unroll
        for (int c = 0; c < 4; ++c) {
            int n = col0 + c * 16 + (lane & 15);
            out[(size_t)m * N + n] = acc[c][v] + bias[n];
        }
    }
}

// ---- fused LSTM step: recurrent 4-gate GEMM + pointwise update ------------
// grid = 32 blocks (32 h-columns each), block = 8 waves.
// wave: (wave&3)->16 batch rows, (wave>>2)->16 h-cols; 4 gate accumulators.
__global__ __launch_bounds__(256) void lstm_step_kernel(
    const unsigned short* __restrict__ Hin,   // [64,1024] bf16 (H_{t-1})
    const unsigned short* __restrict__ WhT,   // [4096,1024] bf16 (gate-major, transposed)
    const float* __restrict__ Xp_t,           // [64,4096] fp32 (precomputed + bias)
    float* __restrict__ Cst,                  // [64,1024] fp32 (in/out)
    unsigned short* __restrict__ Hbf_out,     // [64,1024] bf16 (H_t, = Hs_all+t)
    float* __restrict__ Hf32_out)             // [64,1024] fp32 (for final state)
{
    const int lane  = threadIdx.x & 31;
    const int wave  = threadIdx.x >> 5;
    const int m0    = (wave & 3) * 16;
    const int n0    = blockIdx.x * 32 + (wave >> 2) * 16;
    const int mA    = lane & 15;
    const int khalf = lane >> 4;
    const int nB    = lane & 15;
    const int klo   = (lane >> 4) << 3;

    v8f acc[4] = {{}, {}, {}, {}};
    const unsigned short* Arow = Hin + (size_t)(m0 + mA) * 1024 + khalf * 8;
    const unsigned short* Bp[4];
    #pragma unroll
    for (int g = 0; g < 4; ++g)
        Bp[g] = WhT + (size_t)(g * 1024 + n0 + nB) * 1024 + klo;

    Frag a_c, b_c[4];
    a_c.u[0] = *(const uint4*)(Arow);
    a_c.u[1] = *(const uint4*)(Arow + 16);
    #pragma unroll
    for (int g = 0; g < 4; ++g) {
        b_c[g].u[0] = *(const uint4*)(Bp[g]);
        b_c[g].u[1] = *(const uint4*)(Bp[g] + 16);
    }

    #pragma unroll 2
    for (int kb = 32; kb < 1024; kb += 32) {
        Frag a_n, b_n[4];
        a_n.u[0] = *(const uint4*)(Arow + kb);
        a_n.u[1] = *(const uint4*)(Arow + kb + 16);
        #pragma unroll
        for (int g = 0; g < 4; ++g) {
            b_n[g].u[0] = *(const uint4*)(Bp[g] + kb);
            b_n[g].u[1] = *(const uint4*)(Bp[g] + kb + 16);
        }
        #pragma unroll
        for (int g = 0; g < 4; ++g)
            acc[g] = __builtin_amdgcn_wmma_f32_16x16x32_bf16(
                false, a_c.v, false, b_c[g].v, (short)0, acc[g], false, false);
        a_c = a_n;
        #pragma unroll
        for (int g = 0; g < 4; ++g) b_c[g] = b_n[g];
    }
    #pragma unroll
    for (int g = 0; g < 4; ++g)
        acc[g] = __builtin_amdgcn_wmma_f32_16x16x32_bf16(
            false, a_c.v, false, b_c[g].v, (short)0, acc[g], false, false);

    #pragma unroll
    for (int v = 0; v < 8; ++v) {
        int m = m0 + v + ((lane >> 4) << 3);
        int n = n0 + (lane & 15);
        size_t xrow = (size_t)m * 4096;
        float pi = acc[0][v] + Xp_t[xrow +        n];
        float pf = acc[1][v] + Xp_t[xrow + 1024 + n];
        float po = acc[2][v] + Xp_t[xrow + 2048 + n];
        float pc = acc[3][v] + Xp_t[xrow + 3072 + n];
        float ig = sigmoidf(pi), fg = sigmoidf(pf), og = sigmoidf(po);
        float ct = tanh_fast(pc);
        size_t idx = (size_t)m * 1024 + n;
        float cn = fg * Cst[idx] + ig * ct;
        Cst[idx] = cn;
        float hn = og * tanh_fast(cn);
        Hf32_out[idx] = hn;
        Hbf_out[idx]  = f2bf(hn);
    }
}

// ---------------------------------------------------------------------------
extern "C" void kernel_launch(void* const* d_in, const int* in_sizes, int n_in,
                              void* d_out, int out_size, void* d_ws, size_t ws_size,
                              hipStream_t stream)
{
    constexpr int T = 256, Bz = 64, D = 1024;
    constexpr size_t TB = (size_t)T * Bz;          // 16384
    constexpr size_t G4 = 4 * (size_t)D;           // 4096

    const float* X   = (const float*)d_in[0];
    const float* H0  = (const float*)d_in[1];
    const float* C0  = (const float*)d_in[2];
    const float* Wx[4] = { (const float*)d_in[3],  (const float*)d_in[6],
                           (const float*)d_in[9],  (const float*)d_in[12] };
    const float* Wh[4] = { (const float*)d_in[4],  (const float*)d_in[7],
                           (const float*)d_in[10], (const float*)d_in[13] };
    const float* bg[4] = { (const float*)d_in[5],  (const float*)d_in[8],
                           (const float*)d_in[11], (const float*)d_in[14] };
    const float* Whq = (const float*)d_in[15];
    const float* bq  = (const float*)d_in[16];
    float* out = (float*)d_out;

    // workspace carve-up (256B aligned)
    char* ws = (char*)d_ws;
    size_t off = 0;
    auto carve = [&](size_t bytes) -> void* {
        void* p = ws + off; off += (bytes + 255) & ~(size_t)255; return p;
    };
    unsigned short* Xbf  = (unsigned short*)carve(TB * D * 2);        //  32 MB
    unsigned short* WxT  = (unsigned short*)carve(G4 * D * 2);        //   8 MB
    unsigned short* WhT  = (unsigned short*)carve(G4 * D * 2);        //   8 MB
    unsigned short* WqT  = (unsigned short*)carve((size_t)D * D * 2); //   2 MB
    float*          Xp   = (float*)carve(TB * G4 * 4);                // 256 MB
    unsigned short* Hs   = (unsigned short*)carve(TB * D * 2);        //  32 MB
    unsigned short* H0bf = (unsigned short*)carve((size_t)Bz * D * 2);
    float*          Hf32 = (float*)carve((size_t)Bz * D * 4);
    float*          Cbuf = (float*)carve((size_t)Bz * D * 4);
    float*          bcat = (float*)carve(G4 * 4);

    // -------- phase 1: conversions / transposes ---------------------------
    {
        size_t n8 = TB * D / 8;
        f32_to_bf16_kernel<<<dim3((unsigned)((n8 + 255) / 256)), 256, 0, stream>>>(X, Xbf, n8);
    }
    dim3 tb(32, 8), tg(D / 32, D / 32);
    for (int g = 0; g < 4; ++g) {
        transpose_bf16_kernel<<<tg, tb, 0, stream>>>(Wx[g], WxT + (size_t)g * D * D, D, D);
        transpose_bf16_kernel<<<tg, tb, 0, stream>>>(Wh[g], WhT + (size_t)g * D * D, D, D);
    }
    transpose_bf16_kernel<<<tg, tb, 0, stream>>>(Whq, WqT, D, D);
    init_state_kernel<<<(Bz * D) / 256, 256, 0, stream>>>(H0, C0, H0bf, Cbuf, Hf32, Bz * D);
    build_bias_kernel<<<(int)(G4 / 256), 256, 0, stream>>>(bg[0], bg[1], bg[2], bg[3], bcat);

    // -------- phase 2: Xp = X @ Wx + b  (16384 x 4096, K=1024) ------------
    gemm_bf16_kernel<<<dim3((unsigned)(G4 / 256), (unsigned)(TB / 32)), 256, 0, stream>>>(
        Xbf, WxT, bcat, Xp, (int)G4, D);

    // -------- phase 3: serial recurrence ----------------------------------
    for (int t = 0; t < T; ++t) {
        const unsigned short* Hin = (t == 0) ? H0bf : (Hs + (size_t)(t - 1) * Bz * D);
        lstm_step_kernel<<<D / 32, 256, 0, stream>>>(
            Hin, WhT, Xp + (size_t)t * Bz * G4, Cbuf,
            Hs + (size_t)t * Bz * D, Hf32);
    }

    // -------- phase 4: outputs = H_all @ Whq + bq (16384 x 1024) ----------
    gemm_bf16_kernel<<<dim3((unsigned)(D / 256), (unsigned)(TB / 32)), 256, 0, stream>>>(
        Hs, WqT, bq, out, D, D);

    // -------- final states into d_out tail --------------------------------
    size_t yElems = TB * D;
    hipMemcpyAsync(out + yElems,           Hf32, (size_t)Bz * D * sizeof(float),
                   hipMemcpyDeviceToDevice, stream);
    hipMemcpyAsync(out + yElems + Bz * D,  Cbuf, (size_t)Bz * D * sizeof(float),
                   hipMemcpyDeviceToDevice, stream);
}
